// VectorQuantizerEMA_50337016709433
// MI455X (gfx1250) — compile-verified, matrix operand order checked by reading, not run on hardware
//
#include <hip/hip_runtime.h>
#include <hip/hip_bf16.h>

// ---------------------------------------------------------------------------
// VQ-VAE EMA quantizer for MI455X (gfx1250, wave32, WMMA).
// Heavy op: 65536x512x256 distance GEMM -> v_wmma_f32_16x16x32_bf16 with the
// bf16 codebook staged once per workgroup into LDS via async global->LDS.
// ---------------------------------------------------------------------------

typedef __bf16 bf16_t;
typedef __attribute__((ext_vector_type(16))) __bf16 v16bf;
typedef __attribute__((ext_vector_type(8)))  __bf16 v8bf;
typedef __attribute__((ext_vector_type(8)))  float  v8f;

#define NPOS      65536      // 16 * 16*16*16
#define KCODES    512
#define CDIM      256
#define PSPATIAL  4096       // 16*16*16
#define TOTELEM   16777216   // 16*256*4096
#define CB_PITCH  264        // bf16 elems per LDS codebook row (528 B, bank-safe)
#define CB_LDS_BYTES (KCODES * CB_PITCH * 2)   // 270336 B < 320 KB WGP LDS

// ---------------------------------------------------------------------------
// Pass 1: transpose z_e [B,C,P] -> flat [N,C] (fp32 + bf16), LDS 32x32 tiles.
// ---------------------------------------------------------------------------
__global__ void vq_transpose_kernel(const float* __restrict__ z,
                                    float* __restrict__ flatF,
                                    bf16_t* __restrict__ flatB) {
  __shared__ float tile[32][33];
  const int blk  = blockIdx.x;
  const int pblk = blk & 127;          // 4096/32
  const int cblk = (blk >> 7) & 7;     // 256/32
  const int b    = blk >> 10;          // 16
  const int tx = threadIdx.x & 31;
  const int ty = threadIdx.x >> 5;     // 0..7

  const float* src = z + ((size_t)b * CDIM + (size_t)cblk * 32) * PSPATIAL
                       + (size_t)pblk * 32;
#pragma unroll
  for (int i = 0; i < 4; ++i) {
    const int c = ty + 8 * i;
    tile[c][tx] = src[(size_t)c * PSPATIAL + tx];
  }
  __syncthreads();

  const size_t outBase = ((size_t)b * PSPATIAL + (size_t)pblk * 32) * CDIM
                       + (size_t)cblk * 32;
#pragma unroll
  for (int i = 0; i < 4; ++i) {
    const int p = ty + 8 * i;
    const float v = tile[tx][p];
    flatF[outBase + (size_t)p * CDIM + tx] = v;
    flatB[outBase + (size_t)p * CDIM + tx] = (bf16_t)v;
  }
}

// ---------------------------------------------------------------------------
// Pass 2: codebook -> bf16 copy + per-row squared norm.
// ---------------------------------------------------------------------------
__global__ void vq_cb_prep_kernel(const float* __restrict__ cb,
                                  bf16_t* __restrict__ cbB,
                                  float* __restrict__ e_sq) {
  __shared__ float red[256];
  const int k = blockIdx.x;
  const int c = threadIdx.x;
  const float v = cb[k * CDIM + c];
  cbB[k * CDIM + c] = (bf16_t)v;
  red[c] = v * v;
  __syncthreads();
  for (int s = 128; s > 0; s >>= 1) {
    if (c < s) red[c] += red[c + s];
    __syncthreads();
  }
  if (c == 0) e_sq[k] = red[0];
}

// ---------------------------------------------------------------------------
// Pass 3: distances + argmin via WMMA.
// Codebook (bf16) is staged into LDS once per block with async global->LDS
// copies; each of the 8 waves then runs 4 M-tiles of register-A x LDS-B WMMA.
// A/B 16-bit WMMA layout: lanes 0-15 take K offsets {0..7,16..23},
// lanes 16-31 take {8..15,24..31} inside each 32-wide K chunk.
// ---------------------------------------------------------------------------
__device__ __forceinline__ v16bf vq_mk16(v8bf lo, v8bf hi) {
  v16bf r;
#pragma unroll
  for (int i = 0; i < 8; ++i) { r[i] = lo[i]; r[i + 8] = hi[i]; }
  return r;
}

__global__ void vq_argmin_kernel(const bf16_t* __restrict__ flatB,
                                 const bf16_t* __restrict__ cbB,
                                 const float* __restrict__ e_sq,
                                 int* __restrict__ idx_out) {
  extern __shared__ bf16_t lds_cb[];   // KCODES rows * CB_PITCH bf16

  // ---- Stage 512x256 bf16 codebook into LDS (async, padded pitch) ----
  {
    const unsigned lds_base = (unsigned)(uintptr_t)(void*)lds_cb;
    const unsigned long long gbase = (unsigned long long)(uintptr_t)cbB;
    const unsigned t = threadIdx.x;                // 256 threads
#pragma unroll 4
    for (unsigned i = 0; i < 64; ++i) {            // 16384 x 16B chunks total
      const unsigned idx16 = t + 256u * i;
      const unsigned row   = idx16 >> 5;           // 32 chunks per 512B row
      const unsigned chunk = idx16 & 31u;
      const unsigned goff  = row * 512u + chunk * 16u;
      const unsigned loff  = lds_base + row * (CB_PITCH * 2u) + chunk * 16u;
      asm volatile("global_load_async_to_lds_b128 %0, %1, %2 offset:0"
                   :: "v"(loff), "v"(goff), "s"(gbase)
                   : "memory");
    }
    asm volatile("s_wait_asynccnt 0" ::: "memory");
  }
  __syncthreads();

  const int wave  = threadIdx.x >> 5;             // 0..7
  const int lane  = threadIdx.x & 31;
  const int l16   = lane & 15;
  const int khalf = lane >> 4;                    // 0/1 lane-group

  for (int mtile = blockIdx.x * 8 + wave; mtile < NPOS / 16;
       mtile += gridDim.x * 8) {
    const int m0 = mtile * 16;

    // Load A tile: rows m0..m0+15, C=256 as 8 chunks of K=32.
    v16bf A[8];
    const bf16_t* arow = flatB + (size_t)(m0 + l16) * CDIM;
#pragma unroll
    for (int ck = 0; ck < 8; ++ck) {
      const bf16_t* p = arow + ck * 32 + khalf * 8;
      v8bf lo = *(const v8bf*)(p);
      v8bf hi = *(const v8bf*)(p + 16);
      A[ck] = vq_mk16(lo, hi);
    }

    float bestV[8];
    int   bestI[8];
#pragma unroll
    for (int r = 0; r < 8; ++r) { bestV[r] = 3.4e38f; bestI[r] = 0; }

    for (int t = 0; t < 32; ++t) {                // 32 tiles of 16 codes
      const int code = t * 16 + l16;              // this lane's B column
      const bf16_t* brow = lds_cb + (size_t)code * CB_PITCH;
      v8f acc = {0.f, 0.f, 0.f, 0.f, 0.f, 0.f, 0.f, 0.f};
#pragma unroll
      for (int ck = 0; ck < 8; ++ck) {
        const bf16_t* p = brow + ck * 32 + khalf * 8;
        v8bf lo = *(const v8bf*)(p);              // ds_load_b128
        v8bf hi = *(const v8bf*)(p + 16);         // ds_load_b128
        v16bf B = vq_mk16(lo, hi);
        acc = __builtin_amdgcn_wmma_f32_16x16x32_bf16(
            /*neg_a=*/false, A[ck], /*neg_b=*/false, B,
            /*c_mod=*/(short)0, acc, /*reuse_a=*/false, /*reuse_b=*/false);
      }
      const float es = e_sq[code];
#pragma unroll
      for (int r = 0; r < 8; ++r) {
        const float v = es - 2.0f * acc[r];       // ||e||^2 - 2 z.e
        if (v < bestV[r]) { bestV[r] = v; bestI[r] = code; }
      }
    }

    // Cross-lane argmin within each 16-lane group (C/D layout:
    // lanes 0-15 slot r -> row r ; lanes 16-31 slot r -> row 8+r).
#pragma unroll
    for (int r = 0; r < 8; ++r) {
      float v = bestV[r];
      int   i = bestI[r];
#pragma unroll
      for (int m = 8; m >= 1; m >>= 1) {
        const float ov = __shfl_xor(v, m, 32);
        const int   oi = __shfl_xor(i, m, 32);
        if (ov < v || (ov == v && oi < i)) { v = ov; i = oi; }
      }
      if (l16 == 0) idx_out[m0 + khalf * 8 + r] = i;
    }
  }
}

// ---------------------------------------------------------------------------
// Pass 4: z_q gather back to [B,C,P] + per-block loss partials.
// ---------------------------------------------------------------------------
__global__ void vq_zq_kernel(const float* __restrict__ z,
                             const float* __restrict__ cb,
                             const int* __restrict__ idx,
                             float* __restrict__ zq_out,
                             float* __restrict__ partials) {
  __shared__ float red[256];
  const size_t tid = (size_t)blockIdx.x * 256 + threadIdx.x;  // [b,c,p] flat
  const int p = (int)(tid & 4095);
  const int c = (int)((tid >> 12) & 255);
  const int b = (int)(tid >> 20);
  const int n = b * PSPATIAL + p;
  const int k = idx[n];
  const float q  = cb[k * CDIM + c];
  const float zv = z[tid];
  zq_out[tid] = q;                       // straight-through value == z_q
  float d = zv - q;
  red[threadIdx.x] = d * d;
  __syncthreads();
  for (int s = 128; s > 0; s >>= 1) {
    if (threadIdx.x < s) red[threadIdx.x] += red[threadIdx.x + s];
    __syncthreads();
  }
  if (threadIdx.x == 0) partials[blockIdx.x] = red[0];
}

// ---------------------------------------------------------------------------
// Pass 5: indices as float + LDS-preaggregated histogram.
// ---------------------------------------------------------------------------
__global__ void vq_stats_kernel(const int* __restrict__ idx,
                                float* __restrict__ idx_f_out,
                                int* __restrict__ counts) {
  __shared__ int h[KCODES];
  h[threadIdx.x]       = 0;
  h[threadIdx.x + 256] = 0;
  __syncthreads();
  const int n = blockIdx.x * 256 + threadIdx.x;
  const int k = idx[n];
  idx_f_out[n] = (float)k;
  atomicAdd(&h[k], 1);
  __syncthreads();
  if (h[threadIdx.x])       atomicAdd(&counts[threadIdx.x],       h[threadIdx.x]);
  if (h[threadIdx.x + 256]) atomicAdd(&counts[threadIdx.x + 256], h[threadIdx.x + 256]);
}

// ---------------------------------------------------------------------------
// Pass 6: segment-sum of flat rows into embed_ws (scatter-add).
// ---------------------------------------------------------------------------
__global__ void vq_scatter_kernel(const float* __restrict__ flatF,
                                  const int* __restrict__ idx,
                                  float* __restrict__ embed_ws) {
  const int n = blockIdx.x;          // 65536
  const int c = threadIdx.x;         // 256
  const int k = idx[n];              // uniform within block
  atomicAdd(&embed_ws[k * CDIM + c], flatF[(size_t)n * CDIM + c]);
}

// ---------------------------------------------------------------------------
// Pass 7: deterministic final loss reduction.
// ---------------------------------------------------------------------------
__global__ void vq_loss_reduce_kernel(const float* __restrict__ partials,
                                      float* __restrict__ loss_out) {
  __shared__ float red[256];
  float s = 0.f;
  for (int i = threadIdx.x; i < 65536; i += 256) s += partials[i];
  red[threadIdx.x] = s;
  __syncthreads();
  for (int st = 128; st > 0; st >>= 1) {
    if (threadIdx.x < st) red[threadIdx.x] += red[threadIdx.x + st];
    __syncthreads();
  }
  if (threadIdx.x == 0)
    loss_out[0] = 0.5f * red[0] / (float)TOTELEM;  // COMMITMENT_COST * mean
}

// ---------------------------------------------------------------------------
// Pass 8: EMA cluster-size update + total n.
// ---------------------------------------------------------------------------
__global__ void vq_ema_cs_kernel(const float* __restrict__ ema_cs,
                                 const int* __restrict__ counts,
                                 float* __restrict__ new_cs_out,
                                 float* __restrict__ n_out) {
  __shared__ float red[KCODES];
  const int k = threadIdx.x;  // 512
  const float v = 0.99f * ema_cs[k] + 0.01f * (float)counts[k];
  new_cs_out[k] = v;
  red[k] = v;
  __syncthreads();
  for (int s = 256; s > 0; s >>= 1) {
    if (k < s) red[k] += red[k + s];
    __syncthreads();
  }
  if (k == 0) n_out[0] = red[0];
}

// ---------------------------------------------------------------------------
// Pass 9: EMA embed-sum update + Laplace-smoothed codebook.
// ---------------------------------------------------------------------------
__global__ void vq_ema_embed_kernel(const float* __restrict__ ema_es,
                                    const float* __restrict__ embed_ws,
                                    const float* __restrict__ new_cs,
                                    const float* __restrict__ n_ptr,
                                    float* __restrict__ new_es_out,
                                    float* __restrict__ new_cb_out) {
  const int i = blockIdx.x * 256 + threadIdx.x;  // 131072
  const int k = i >> 8;
  const float nes = 0.99f * ema_es[i] + 0.01f * embed_ws[i];
  new_es_out[i] = nes;
  const float n = n_ptr[0];
  const float smoothed = (new_cs[k] + 1e-5f) / (n + (float)KCODES * 1e-5f) * n;
  new_cb_out[i] = nes / smoothed;
}

// ---------------------------------------------------------------------------
extern "C" void kernel_launch(void* const* d_in, const int* in_sizes, int n_in,
                              void* d_out, int out_size, void* d_ws, size_t ws_size,
                              hipStream_t stream) {
  (void)in_sizes; (void)n_in; (void)out_size; (void)ws_size;
  const float* z      = (const float*)d_in[0];
  const float* cb     = (const float*)d_in[1];
  const float* ema_cs = (const float*)d_in[2];
  const float* ema_es = (const float*)d_in[3];

  float* out      = (float*)d_out;
  float* out_zq   = out;               // 16777216
  float* out_loss = out + 16777216;    // 1
  float* out_idx  = out + 16777217;    // 65536
  float* out_cb   = out + 16842753;    // 131072
  float* out_cs   = out + 16973825;    // 512
  float* out_es   = out + 16974337;    // 131072

  char* ws = (char*)d_ws;
  float*  flatF    = (float*)(ws);                      // 64 MB
  bf16_t* flatB    = (bf16_t*)(ws + 67108864);          // 32 MB
  bf16_t* cbB      = (bf16_t*)(ws + 100663296);         // 256 KB
  float*  e_sq     = (float*)(ws + 100925440);          // 2 KB
  int*    idx      = (int*)(ws + 100927488);            // 256 KB
  int*    counts   = (int*)(ws + 101189632);            // 2 KB
  float*  embed_ws = (float*)(ws + 101191680);          // 512 KB
  float*  partials = (float*)(ws + 101715968);          // 256 KB
  float*  n_val    = (float*)(ws + 101978112);          // 4 B

  hipMemsetAsync(counts, 0, KCODES * sizeof(int), stream);
  hipMemsetAsync(embed_ws, 0, KCODES * CDIM * sizeof(float), stream);

  vq_transpose_kernel<<<16384, 256, 0, stream>>>(z, flatF, flatB);
  vq_cb_prep_kernel<<<KCODES, 256, 0, stream>>>(cb, cbB, e_sq);
  vq_argmin_kernel<<<128, 256, CB_LDS_BYTES, stream>>>(flatB, cbB, e_sq, idx);
  vq_zq_kernel<<<65536, 256, 0, stream>>>(z, cb, idx, out_zq, partials);
  vq_stats_kernel<<<256, 256, 0, stream>>>(idx, out_idx, counts);
  vq_scatter_kernel<<<NPOS, 256, 0, stream>>>(flatF, idx, embed_ws);
  vq_loss_reduce_kernel<<<1, 256, 0, stream>>>(partials, out_loss);
  vq_ema_cs_kernel<<<1, KCODES, 0, stream>>>(ema_cs, counts, out_cs, n_val);
  vq_ema_embed_kernel<<<512, 256, 0, stream>>>(ema_es, embed_ws, out_cs, n_val,
                                               out_es, out_cb);
}
